// TransducerJoint_3582002725423
// MI455X (gfx1250) — matrix operand order repaired and checked by reading, not compiled, not run on hardware
//
#include <hip/hip_runtime.h>
#include <cmath>

typedef __attribute__((ext_vector_type(16))) __bf16 v16bf;
typedef __attribute__((ext_vector_type(8)))  float  v8f;

#define B_    8
#define T_    200
#define U_    100
#define DENC  512
#define DPRED 640
#define J_    640
#define V_    1024
#define KSTEPS (J_ / 32)        // 20 k-steps of 32
#define NTILES (V_ / 16)        // 64 n-tiles of 16
#define UTILES ((U_ + 15) / 16) // 7 u-tiles (last partial)

// Fast hardware tanh (V_TANH_F32, 1 TRANS op) when the toolchain exposes it.
#if __has_builtin(__builtin_amdgcn_tanhf)
  #define FAST_TANH(x) __builtin_amdgcn_tanhf(x)
#elif __has_builtin(__builtin_amdgcn_tanh_f32)
  #define FAST_TANH(x) __builtin_amdgcn_tanh_f32(x)
#else
  #define FAST_TANH(x) tanhf(x)
#endif

// ---------------------------------------------------------------------------
// Small projection GEMM (fp32): Y[m,n] = sum_k X[m,k]*W[k,n] + bias[n]
// <1% of total FLOPs; plain FMA kernel, coalesced over n.
// ---------------------------------------------------------------------------
__global__ void __launch_bounds__(256) proj_kernel(
    const float* __restrict__ X, const float* __restrict__ W,
    const float* __restrict__ bias, float* __restrict__ Y,
    int M, int K, int N)
{
    int idx = blockIdx.x * blockDim.x + threadIdx.x;
    if (idx >= M * N) return;
    int m = idx / N;
    int n = idx - m * N;
    const float* x = X + (long)m * K;
    const float* w = W + n;
    float s = bias[n];
#pragma unroll 8
    for (int k = 0; k < K; ++k)
        s = fmaf(x[k], w[(long)k * N], s);
    Y[idx] = s;
}

// ---------------------------------------------------------------------------
// Pack W_out (fp32 row-major [J,V]) -> bf16 B-fragments.
// B layout packed here per 32x16 (KxN) tile (contiguous K, A uses the same order):
//   lane n   (0..15): K = 0..15  of column n, packed 2 per VGPR
//   lane n+16       : K = 16..31 of column n
// Flat: Bpk[((ntile*KSTEPS + kstep)*32 + lane)*16 + slot]
// ---------------------------------------------------------------------------
__global__ void __launch_bounds__(256) pack_wout_kernel(
    const float* __restrict__ W, __bf16* __restrict__ Bpk)
{
    int idx = blockIdx.x * blockDim.x + threadIdx.x;
    if (idx >= J_ * V_) return;
    int k = idx / V_;
    int v = idx - k * V_;
    int ntile = v >> 4, n = v & 15;
    int kstep = k >> 5, kk = k & 31;
    int lane = n + ((kk >= 16) ? 16 : 0);
    int slot = kk & 15;
    int dest = ((ntile * KSTEPS + kstep) * 32 + lane) * 16 + slot;
    Bpk[dest] = (__bf16)W[idx];
}

// ---------------------------------------------------------------------------
// Fused joint + vocab GEMM.
// Block: 256 threads (8 waves). One block = 16 rows (fixed b,t; u0..u0+15)
// x all 1024 vocab columns. Wave w computes n-tiles [w*8, w*8+8).
// Double-buffered A tile in LDS (1 barrier per K-step); all 8 B fragments
// loaded into registers up-front so WMMA overlaps the load tail.
// ---------------------------------------------------------------------------
__global__ void __launch_bounds__(256) joint_gemm_kernel(
    const float* __restrict__ enc_out,   // [B,T,J] f32
    const float* __restrict__ pred_out,  // [B,U,J] f32
    const __bf16* __restrict__ Bpk,      // packed W_out bf16
    const float* __restrict__ b_out,     // [V] f32
    float* __restrict__ out)             // [B,T,U,V] f32
{
    __shared__ __align__(32) __bf16 Atile[2][512]; // 2 x (16x32 bf16 fragment)

    int blk = blockIdx.x;
    int ut  = blk % UTILES;
    int bt  = blk / UTILES;          // 0 .. B_*T_-1
    int b   = bt / T_;
    int u0  = ut * 16;

    const float* encRow   = enc_out  + (long)bt * J_;
    const float* predBase = pred_out + (long)b * U_ * J_;

    int tid  = threadIdx.x;
    int lane = tid & 31;
    int wave = tid >> 5;

    // ---- A-tile build assignment: thread -> (row m, kk pair) ----
    int m_b  = tid >> 4;            // 0..15 : row within tile
    int kk_b = (tid & 15) << 1;     // 0,2,...,30 : two consecutive K
    int ub   = u0 + m_b;
    if (ub > U_ - 1) ub = U_ - 1;   // clamp OOB rows (outputs masked later)
    const float* predRow = predBase + (long)ub * J_;

    // A-fragment LDS slot (contiguous K, matching packed B):
    // lane = m + 16*(kk>=16); slot = kk&15
    int grp     = kk_b >> 3;
    int lds_off = (m_b + 16 * (grp >> 1)) * 16 + (kk_b & 15);

    v8f acc[8];
#pragma unroll
    for (int i = 0; i < 8; ++i)
#pragma unroll
        for (int j = 0; j < 8; ++j)
            acc[i][j] = 0.0f;

    const __bf16* bbase = Bpk + (long)(wave * 8) * KSTEPS * 512 + lane * 16;

    // Build A tile for kstep 0 into buffer 0.
    {
        float x0 = encRow[kk_b]     + predRow[kk_b];
        float x1 = encRow[kk_b + 1] + predRow[kk_b + 1];
        Atile[0][lds_off]     = (__bf16)FAST_TANH(x0);
        Atile[0][lds_off + 1] = (__bf16)FAST_TANH(x1);
    }
    __syncthreads();

    for (int ks = 0; ks < KSTEPS; ++ks) {
        // Consume current A fragment.
        v16bf afrag = *(const v16bf*)&Atile[ks & 1][lane * 16];

        // Build next K-step's A tile into the other buffer (block-uniform branch).
        if (ks + 1 < KSTEPS) {
            int k0 = (ks + 1) * 32;
            float x0 = encRow[k0 + kk_b]     + predRow[k0 + kk_b];
            float x1 = encRow[k0 + kk_b + 1] + predRow[k0 + kk_b + 1];
            Atile[(ks + 1) & 1][lds_off]     = (__bf16)FAST_TANH(x0);
            Atile[(ks + 1) & 1][lds_off + 1] = (__bf16)FAST_TANH(x1);
        }

        // Load all 8 B fragments (one clause of 16 x global_load_b128),
        // then issue the 8 WMMAs so the load tail overlaps matrix math.
        v16bf bf[8];
#pragma unroll
        for (int nt = 0; nt < 8; ++nt)
            bf[nt] = *(const v16bf*)(bbase + (long)(nt * KSTEPS + ks) * 512);
#pragma unroll
        for (int nt = 0; nt < 8; ++nt)
            acc[nt] = __builtin_amdgcn_wmma_f32_16x16x32_bf16(
                false, afrag, false, bf[nt], (short)0, acc[nt], false, false);

        // Single barrier per K-step: publishes next buffer's writes AND
        // retires this buffer's reads before it is overwritten next step.
        __syncthreads();
    }

    // ---- epilogue: bias + store (C/D layout: VGPR g -> M=g (lanes 0-15),
    // M=8+g (lanes 16-31); N = lane&15) ----
    int ncol  = lane & 15;
    int mbase = (lane >> 4) << 3;   // 0 or 8
    long rowbase = (long)bt * U_ + u0;
#pragma unroll
    for (int nt = 0; nt < 8; ++nt) {
        int n = (wave * 8 + nt) * 16 + ncol;
        float bias = b_out[n];
#pragma unroll
        for (int g = 0; g < 8; ++g) {
            int u = u0 + mbase + g;
            if (u < U_)
                out[(rowbase + mbase + g) * V_ + n] = acc[nt][g] + bias;
        }
    }
}

// ---------------------------------------------------------------------------
// Launch
// ---------------------------------------------------------------------------
extern "C" void kernel_launch(void* const* d_in, const int* in_sizes, int n_in,
                              void* d_out, int out_size, void* d_ws, size_t ws_size,
                              hipStream_t stream)
{
    const float* enc    = (const float*)d_in[0];
    const float* pred   = (const float*)d_in[1];
    const float* W_enc  = (const float*)d_in[2];
    const float* b_enc  = (const float*)d_in[3];
    const float* W_pred = (const float*)d_in[4];
    const float* b_pred = (const float*)d_in[5];
    const float* W_out  = (const float*)d_in[6];
    const float* b_out  = (const float*)d_in[7];
    float* out = (float*)d_out;

    // Workspace carve-up (all 256B-aligned offsets):
    //   [0, 1310720)                 : Bpk      bf16  J_*V_ = 655360 elems
    //   [1310720, 5406720)           : enc_out  f32   B*T*J = 1,024,000
    //   [5406720, 7454720)           : pred_out f32   B*U*J = 512,000
    __bf16* Bpk      = (__bf16*)d_ws;
    float*  enc_out  = (float*)((char*)d_ws + 1310720);
    float*  pred_out = (float*)((char*)d_ws + 1310720 + 4096000);

    // 1) enc projection: [B*T, DENC] x [DENC, J]
    {
        int M = B_ * T_, K = DENC, N = J_;
        int total = M * N;
        proj_kernel<<<(total + 255) / 256, 256, 0, stream>>>(
            enc, W_enc, b_enc, enc_out, M, K, N);
    }
    // 2) pred projection: [B*U, DPRED] x [DPRED, J]
    {
        int M = B_ * U_, K = DPRED, N = J_;
        int total = M * N;
        proj_kernel<<<(total + 255) / 256, 256, 0, stream>>>(
            pred, W_pred, b_pred, pred_out, M, K, N);
    }
    // 3) pack W_out into bf16 B-fragments
    {
        int total = J_ * V_;
        pack_wout_kernel<<<(total + 255) / 256, 256, 0, stream>>>(W_out, Bpk);
    }
    // 4) fused tanh-joint + vocab GEMM (WMMA bf16)
    {
        int blocks = B_ * T_ * UTILES; // 11200
        joint_gemm_kernel<<<blocks, 256, 0, stream>>>(
            enc_out, pred_out, Bpk, b_out, out);
    }
}
